// FractalMambaModel_80805514707212
// MI455X (gfx1250) — matrix operand order repaired and checked by reading, not compiled, not run on hardware
//
#include <hip/hip_runtime.h>

typedef __attribute__((ext_vector_type(16))) __bf16 v16bf;
typedef __attribute__((ext_vector_type(8)))  float  v8f;
typedef unsigned short u16;
typedef unsigned int   u32;

#define NLAYER 8
#define HD     768
#define ED     1536
#define NS     16
#define RD     4
#define KC     4
#define CF     3
#define BATCH  2
#define LSEQ   2048
#define MROWS  (BATCH*LSEQ)   /* 4096 */
#define DBCW   (RD + 2*NS)    /* 36 */

__device__ __forceinline__ u16 f2b(float f){
  u32 u = __float_as_uint(f);
  u32 r = (u + 0x7FFFu + ((u >> 16) & 1u)) >> 16;
  return (u16)r;
}
__device__ __forceinline__ float sigmoidf_(float x){
  return __builtin_amdgcn_rcpf(1.0f + __expf(-x));   /* v_rcp_f32, no slow div */
}
__device__ __forceinline__ float siluf_(float x){ return x * sigmoidf_(x); }

/* CDNA5 async global->LDS copy (16B per lane), tracked by ASYNCcnt.
   lds_off is a byte offset into the workgroup LDS allocation (single
   __shared__ array => allocation starts at offset 0).
   lds_base (flat pointer to the shared array) is passed as a dummy
   operand so the allocation ESCAPES into the asm: with the "memory"
   clobber the compiler must then assume the asm writes the LDS array,
   keeping the subsequent ds_load fragments ordered and live.          */
__device__ __forceinline__ void async_ld_b128(u32 lds_off, const void* gaddr,
                                              const void* lds_base){
  asm volatile("global_load_async_to_lds_b128 %0, %1, off"
               :: "v"(lds_off), "v"(gaddr), "v"(lds_base) : "memory");
}
__device__ __forceinline__ void wait_async1(){
  asm volatile("s_wait_asynccnt 0x1" ::: "memory");
}
__device__ __forceinline__ void wait_async0(){
  asm volatile("s_wait_asynccnt 0x0" ::: "memory");
}

/* ------------------------------------------------------------------ */
/* Embedding: hs[b,l,:] = tok_emb[ids[b,l],:] + pos_emb[l,:]           */
/* ------------------------------------------------------------------ */
__global__ __launch_bounds__(256) void k_embed(const int* __restrict__ ids,
                                               const float* __restrict__ tok,
                                               const float* __restrict__ pos,
                                               float* __restrict__ hs)
{
  size_t idx = (size_t)blockIdx.x * blockDim.x + threadIdx.x;
  size_t n = (size_t)MROWS * HD;
  if (idx >= n) return;
  int h   = (int)(idx % HD);
  int row = (int)(idx / HD);
  int l   = row % LSEQ;
  int id  = ids[row];
  hs[idx] = tok[(size_t)id * HD + h] + pos[(size_t)l * HD + h];
}

/* ------------------------------------------------------------------ */
/* LayerNorm over H=768 (3 elems/thread, 256-thread block per row).    */
/* ------------------------------------------------------------------ */
__global__ __launch_bounds__(256) void k_layernorm(const float* __restrict__ X,
                                                   const float* __restrict__ w,
                                                   const float* __restrict__ b,
                                                   u16* __restrict__ outB,
                                                   float* __restrict__ outF)
{
  __shared__ float red[8];
  const int row = blockIdx.x;
  const int tid = threadIdx.x;
  const float* xr = X + (size_t)row * HD;
  float x0 = xr[tid], x1 = xr[tid + 256], x2 = xr[tid + 512];

  float s = x0 + x1 + x2;
  #pragma unroll
  for (int o = 16; o > 0; o >>= 1) s += __shfl_xor(s, o);
  if ((tid & 31) == 0) red[tid >> 5] = s;
  __syncthreads();
  float tot = 0.f;
  #pragma unroll
  for (int i = 0; i < 8; i++) tot += red[i];
  const float mean = tot * (1.0f / HD);
  __syncthreads();

  float d0 = x0 - mean, d1 = x1 - mean, d2 = x2 - mean;
  s = d0*d0 + d1*d1 + d2*d2;
  #pragma unroll
  for (int o = 16; o > 0; o >>= 1) s += __shfl_xor(s, o);
  if ((tid & 31) == 0) red[tid >> 5] = s;
  __syncthreads();
  tot = 0.f;
  #pragma unroll
  for (int i = 0; i < 8; i++) tot += red[i];
  const float rs = rsqrtf(tot * (1.0f / HD) + 1e-5f);

  const size_t base = (size_t)row * HD;
  float d[3] = {d0, d1, d2};
  #pragma unroll
  for (int j = 0; j < 3; j++){
    int col = tid + j * 256;
    float v = d[j] * rs * w[col] + b[col];
    if (outB) outB[base + col] = f2b(v);
    if (outF) outF[base + col] = v;
  }
}

/* ------------------------------------------------------------------ */
/* Pack f32 KxN weight into fragment-contiguous bf16 layout:           */
/* per (kblk=k/32, nblk=n/16) tile, lane (n%16 + 16*(k%32/16)) keeps   */
/* its 8 WMMA dwords back-to-back:                                     */
/*   dst = ((kblk*(N/16)+nblk)*32 + lane)*16 + ((k%16)/2)*2 + (k&1)    */
/* so each lane's B fragment is one 32B chunk, and a 32x64 slab is a   */
/* contiguous 4KB region (perfect for async b128 staging).             */
/* ------------------------------------------------------------------ */
__global__ __launch_bounds__(256) void k_pack(const float* __restrict__ W,
                                              u16* __restrict__ Bp, int Kd, int Nd)
{
  size_t idx = (size_t)blockIdx.x * blockDim.x + threadIdx.x;
  size_t n = (size_t)Kd * Nd;
  if (idx >= n) return;
  int k = (int)(idx / Nd);
  int c = (int)(idx % Nd);
  int kblk = k >> 5, kin = k & 31;
  int nblk = c >> 4, nin = c & 15;
  int lane = nin + ((kin >> 4) << 4);
  int v    = (kin & 15) >> 1;
  int lo   = kin & 1;
  size_t dst = (((size_t)(kblk * (Nd >> 4) + nblk)) * 32 + lane) * 16 + v * 2 + lo;
  Bp[dst] = f2b(W[idx]);
}

/* f32 -> bf16 row-major copy */
__global__ __launch_bounds__(256) void k_f2b(const float* __restrict__ X,
                                             u16* __restrict__ Y, size_t n)
{
  size_t idx = (size_t)blockIdx.x * blockDim.x + threadIdx.x;
  if (idx < n) Y[idx] = f2b(X[idx]);
}

/* ------------------------------------------------------------------ */
/* bf16 WMMA GEMM, LDS-staged B with CDNA5 async-to-LDS double buffer. */
/* Block: 256 thr = 8 waves -> 256(M) x 64(N) tile; wave w owns rows   */
/* [w*32, w*32+32) as two 16-row WMMA tiles x four 16-col tiles        */
/* => 8 v_wmma_f32_16x16x32_bf16 per 32-wide k-step.                   */
/* ------------------------------------------------------------------ */
union FragAB { u32 u[8]; v16bf v; };

#define BM 256
#define BN 64

__global__ __launch_bounds__(256) void k_gemm_wmma(
    const u16* __restrict__ A, const u16* __restrict__ Bp,
    const float* __restrict__ bias, const float* __restrict__ addsrc,
    float* __restrict__ outF, u16* __restrict__ outB,
    int M, int N, int K, int act, int accum)
{
  __shared__ __align__(16) u16 sB[2 * 2048];   /* 2 x 4KB slabs, LDS offset 0 */

  const int tid   = (int)threadIdx.x;
  const int w     = tid >> 5;
  const int lane  = tid & 31;
  const int laneM = lane & 15;
  const int hi    = lane >> 4;

  const int nblk4 = N >> 6;                    /* 64-col blocks */
  const int bm = (int)blockIdx.x / nblk4;
  const int bn = (int)blockIdx.x % nblk4;

  const int   NT16       = N >> 4;
  const size_t slabStride = (size_t)NT16 * 512;        /* u16 per kblk */
  const u16*  gslab = Bp + (size_t)(bn * 4) * 512 + (size_t)tid * 8;

  const u32 off0 = (u32)(tid * 16);
  const u32 off1 = (u32)(4096 + tid * 16);

  int ak[8];
  #pragma unroll
  for (int v = 0; v < 8; v++)
    ak[v] = hi * 16 + 2 * v;

  const size_t arow0 = (size_t)(bm * BM + w * 32 + laneM) * K;
  const size_t arow1 = arow0 + (size_t)16 * K;

  v8f c[8];
  #pragma unroll
  for (int t = 0; t < 8; t++)
    #pragma unroll
    for (int r = 0; r < 8; r++) c[t][r] = 0.0f;

  /* prologue: stage k-block 0 */
  async_ld_b128(off0, gslab, sB);

  const int nk = K >> 5;
  int cur = 0;
  for (int kb = 0; kb < nk; kb++){
    const int kk = kb << 5;
    if (kb + 1 < nk){
      async_ld_b128(cur ? off0 : off1, gslab + (size_t)(kb + 1) * slabStride, sB);
      wait_async1();                            /* older (current) slab done */
    } else {
      wait_async0();
    }
    __syncthreads();                            /* all waves' slab parts in */

    FragAB a0, a1;
    #pragma unroll
    for (int v = 0; v < 8; v++){
      a0.u[v] = *(const u32*)(A + arow0 + kk + ak[v]);
      a1.u[v] = *(const u32*)(A + arow1 + kk + ak[v]);
    }

    const u16* lbuf = &sB[cur * 2048];
    #pragma unroll
    for (int t = 0; t < 4; t++){
      FragAB bfrag;
      const u32* lp = (const u32*)(lbuf + (size_t)(t * 32 + lane) * 16);
      #pragma unroll
      for (int v = 0; v < 8; v++) bfrag.u[v] = lp[v];
      c[t]     = __builtin_amdgcn_wmma_f32_16x16x32_bf16(
                   false, a0.v, false, bfrag.v, (short)0, c[t],     false, false);
      c[4 + t] = __builtin_amdgcn_wmma_f32_16x16x32_bf16(
                   false, a1.v, false, bfrag.v, (short)0, c[4 + t], false, false);
    }
    __syncthreads();                            /* reads done before overwrite */
    cur ^= 1;
  }

  #pragma unroll
  for (int mt = 0; mt < 2; mt++){
    const int rbase = bm * BM + w * 32 + mt * 16 + 8 * hi;
    #pragma unroll
    for (int t = 0; t < 4; t++){
      const int col = bn * BN + t * 16 + laneM;
      #pragma unroll
      for (int r = 0; r < 8; r++){
        const int row = rbase + r;
        const size_t idx = (size_t)row * N + col;
        float v = c[mt * 4 + t][r] + bias[col];
        if (addsrc) v += addsrc[idx];
        if (act) v = siluf_(v);
        if (outF){ if (accum) outF[idx] += v; else outF[idx] = v; }
        if (outB) outB[idx] = f2b(v);
      }
    }
  }
}

/* ------------------------------------------------------------------ */
/* Causal depthwise conv (K=4) on proj[:, :E] + SiLU -> u              */
/* ------------------------------------------------------------------ */
__global__ __launch_bounds__(256) void k_conv_silu(const float* __restrict__ proj,
                                                   const float* __restrict__ cw,
                                                   const float* __restrict__ cb,
                                                   float* __restrict__ U)
{
  size_t idx = (size_t)blockIdx.x * blockDim.x + threadIdx.x;
  if (idx >= (size_t)MROWS * ED) return;
  int e = (int)(idx % ED);
  int m = (int)(idx / ED);
  int l = m % LSEQ;
  int b = m / LSEQ;
  float acc = cb[e];
  #pragma unroll
  for (int j = 0; j < KC; j++){
    int li = l + j - (KC - 1);
    if (li >= 0)
      acc += proj[((size_t)(b * LSEQ + li)) * (2 * ED) + e] * cw[e * KC + j];
  }
  U[idx] = siluf_(acc);
}

/* dbc[m, j] = u[m,:] . xproj_w[:, j]   (j < 36) */
__global__ __launch_bounds__(256) void k_dbc(const float* __restrict__ U,
                                             const float* __restrict__ XW,
                                             float* __restrict__ dbc)
{
  size_t idx = (size_t)blockIdx.x * blockDim.x + threadIdx.x;
  if (idx >= (size_t)MROWS * DBCW) return;
  int j = (int)(idx % DBCW);
  int m = (int)(idx / DBCW);
  const float* ur = U + (size_t)m * ED;
  float s = 0.f;
  for (int k = 0; k < ED; k++) s += ur[k] * XW[(size_t)k * DBCW + j];
  dbc[idx] = s;
}

/* dt[m,e] = softplus(dbc[m,:R] . dt_w[:,e] + dt_b[e]) */
__global__ __launch_bounds__(256) void k_dt(const float* __restrict__ dbc,
                                            const float* __restrict__ DW,
                                            const float* __restrict__ DB,
                                            float* __restrict__ DT)
{
  size_t idx = (size_t)blockIdx.x * blockDim.x + threadIdx.x;
  if (idx >= (size_t)MROWS * ED) return;
  int e = (int)(idx % ED);
  int m = (int)(idx / ED);
  float s = DB[e];
  #pragma unroll
  for (int r = 0; r < RD; r++)
    s += dbc[(size_t)m * DBCW + r] * DW[(size_t)r * ED + e];
  DT[idx] = (s > 20.0f) ? s : log1pf(__expf(s));
}

/* ------------------------------------------------------------------ */
/* Selective scan: each 16-lane half-wave owns one (b,e) channel,      */
/* lane n holds state h_n; y reduced with shfl_xor within the half.    */
/* ------------------------------------------------------------------ */
__global__ __launch_bounds__(256) void k_scan(const float* __restrict__ U,
                                              const float* __restrict__ DT,
                                              const float* __restrict__ dbc,
                                              const float* __restrict__ Alog,
                                              const float* __restrict__ Dv,
                                              float* __restrict__ Y)
{
  const int wave = (int)((blockIdx.x * blockDim.x + threadIdx.x) >> 5);
  const int lane = (int)(threadIdx.x & 31);
  const int sub  = lane >> 4;
  const int n    = lane & 15;
  const int chan = wave * 2 + sub;
  if (chan >= BATCH * ED) return;
  const int b = chan / ED;
  const int e = chan % ED;

  const float An = -__expf(Alog[(size_t)e * NS + n]);
  const float Dn = Dv[e];
  float h = 0.0f;

  for (int l = 0; l < LSEQ; l++){
    const size_t m = (size_t)b * LSEQ + l;
    const float u  = U [m * ED + e];
    const float dt = DT[m * ED + e];
    const float Bv = dbc[m * DBCW + RD + n];
    const float Cv = dbc[m * DBCW + RD + NS + n];
    h = h * __expf(dt * An) + dt * u * Bv;
    float yp = h * Cv;
    yp += __shfl_xor(yp, 1);
    yp += __shfl_xor(yp, 2);
    yp += __shfl_xor(yp, 4);
    yp += __shfl_xor(yp, 8);
    if (n == 0) Y[m * ED + e] = yp + u * Dn;
  }
}

/* y = sigmoid(gate)*scan_y  -> bf16 A-matrix for out_proj GEMM */
__global__ __launch_bounds__(256) void k_gate(const float* __restrict__ proj,
                                              const float* __restrict__ SY,
                                              u16* __restrict__ YB)
{
  size_t idx = (size_t)blockIdx.x * blockDim.x + threadIdx.x;
  if (idx >= (size_t)MROWS * ED) return;
  int e = (int)(idx % ED);
  int m = (int)(idx / ED);
  float g = proj[(size_t)m * (2 * ED) + ED + e];
  YB[idx] = f2b(sigmoidf_(g) * SY[idx]);
}

/* hs += (0.5/C) * acc */
__global__ __launch_bounds__(256) void k_fracadd(float* __restrict__ hs,
                                                 const float* __restrict__ acc)
{
  size_t idx = (size_t)blockIdx.x * blockDim.x + threadIdx.x;
  if (idx < (size_t)MROWS * HD) hs[idx] += (0.5f / (float)CF) * acc[idx];
}

/* ------------------------------------------------------------------ */
extern "C" void kernel_launch(void* const* d_in, const int* in_sizes, int n_in,
                              void* d_out, int out_size, void* d_ws, size_t ws_size,
                              hipStream_t stream)
{
  (void)in_sizes; (void)n_in; (void)out_size; (void)ws_size;
  const int*   ids    = (const int*)  d_in[0];
  const float* tok    = (const float*)d_in[1];
  const float* pos    = (const float*)d_in[2];
  const float* ln_w   = (const float*)d_in[3];
  const float* ln_b   = (const float*)d_in[4];
  const float* in_w   = (const float*)d_in[5];
  const float* in_b   = (const float*)d_in[6];
  const float* conv_w = (const float*)d_in[7];
  const float* conv_b = (const float*)d_in[8];
  const float* xprj_w = (const float*)d_in[9];
  const float* dt_w   = (const float*)d_in[10];
  const float* dt_b   = (const float*)d_in[11];
  const float* A_log  = (const float*)d_in[12];
  const float* Dv     = (const float*)d_in[13];
  const float* out_w  = (const float*)d_in[14];
  const float* out_b  = (const float*)d_in[15];
  const float* frac_w = (const float*)d_in[16];
  const float* frac_b = (const float*)d_in[17];
  const float* fln_w  = (const float*)d_in[18];
  const float* fln_b  = (const float*)d_in[19];
  float* outp = (float*)d_out;

  /* workspace carve-up (256B aligned) */
  char* ws = (char*)d_ws;
  size_t off = 0;
  auto carve = [&](size_t bytes) -> char* {
    char* p = ws + off;
    off = (off + bytes + 255) & ~(size_t)255;
    return p;
  };
  float* hs   = (float*)carve((size_t)MROWS * HD * 4);
  float* proj = (float*)carve((size_t)MROWS * 2 * ED * 4);
  float* U    = (float*)carve((size_t)MROWS * ED * 4);
  float* DT   = (float*)carve((size_t)MROWS * ED * 4);
  float* SY   = (float*)carve((size_t)MROWS * ED * 4);
  float* dbc  = (float*)carve((size_t)MROWS * DBCW * 4);
  float* acc  = (float*)carve((size_t)MROWS * HD * 4);
  u16* xlnb   = (u16*)carve((size_t)MROWS * HD * 2);
  u16* hsb    = (u16*)carve((size_t)MROWS * HD * 2);
  u16* t1b    = (u16*)carve((size_t)MROWS * HD * 2);
  u16* t2b    = (u16*)carve((size_t)MROWS * HD * 2);
  u16* ybb    = (u16*)carve((size_t)MROWS * ED * 2);
  u16* wpin   = (u16*)carve((size_t)HD * 2 * ED * 2);
  u16* wpout  = (u16*)carve((size_t)ED * HD * 2);
  u16* wpf    = (u16*)carve((size_t)HD * HD * 2);

  auto ew = [](size_t n) { return dim3((unsigned)((n + 255) / 256)); };
  auto gemm = [&](const u16* Ab, const u16* Bp, const float* bias,
                  const float* addsrc, float* oF, u16* oB,
                  int M, int N, int K, int act, int accum){
    int blocks = (M / BM) * (N / BN);
    hipLaunchKernelGGL(k_gemm_wmma, dim3(blocks), dim3(256), 0, stream,
                       Ab, Bp, bias, addsrc, oF, oB, M, N, K, act, accum);
  };

  hipLaunchKernelGGL(k_embed, ew((size_t)MROWS * HD), dim3(256), 0, stream, ids, tok, pos, hs);

  for (int l = 0; l < NLAYER; l++){
    /* x = LN(hs) -> bf16 */
    hipLaunchKernelGGL(k_layernorm, dim3(MROWS), dim3(256), 0, stream,
                       hs, ln_w + (size_t)l * HD, ln_b + (size_t)l * HD, xlnb, (float*)nullptr);
    /* proj = x @ in_w + in_b */
    hipLaunchKernelGGL(k_pack, ew((size_t)HD * 2 * ED), dim3(256), 0, stream,
                       in_w + (size_t)l * HD * 2 * ED, wpin, HD, 2 * ED);
    gemm(xlnb, wpin, in_b + (size_t)l * 2 * ED, nullptr, proj, nullptr,
         MROWS, 2 * ED, HD, 0, 0);
    /* u = silu(causal dwconv(xm)) */
    hipLaunchKernelGGL(k_conv_silu, ew((size_t)MROWS * ED), dim3(256), 0, stream,
                       proj, conv_w + (size_t)l * ED * KC, conv_b + (size_t)l * ED, U);
    /* dbc, dt */
    hipLaunchKernelGGL(k_dbc, ew((size_t)MROWS * DBCW), dim3(256), 0, stream,
                       U, xprj_w + (size_t)l * ED * DBCW, dbc);
    hipLaunchKernelGGL(k_dt, ew((size_t)MROWS * ED), dim3(256), 0, stream,
                       dbc, dt_w + (size_t)l * RD * ED, dt_b + (size_t)l * ED, DT);
    /* selective scan */
    {
      int waves = (BATCH * ED) / 2;               /* 1536 waves */
      hipLaunchKernelGGL(k_scan, dim3(waves * 32 / 256), dim3(256), 0, stream,
                         U, DT, dbc, A_log + (size_t)l * ED * NS, Dv + (size_t)l * ED, SY);
    }
    /* y = sigmoid(gate)*scan_y -> bf16 */
    hipLaunchKernelGGL(k_gate, ew((size_t)MROWS * ED), dim3(256), 0, stream, proj, SY, ybb);
    /* hs = y @ out_w + out_b + res */
    hipLaunchKernelGGL(k_pack, ew((size_t)ED * HD), dim3(256), 0, stream,
                       out_w + (size_t)l * ED * HD, wpout, ED, HD);
    gemm(ybb, wpout, out_b + (size_t)l * HD, hs, hs, nullptr, MROWS, HD, ED, 0, 0);
    /* fractal block */
    hipLaunchKernelGGL(k_f2b, ew((size_t)MROWS * HD), dim3(256), 0, stream,
                       hs, hsb, (size_t)MROWS * HD);
    for (int i = 0; i < CF; i++){
      const float* fw = frac_w + ((size_t)l * CF + i) * HD * HD;
      const float* fb = frac_b + ((size_t)l * CF + i) * HD;
      hipLaunchKernelGGL(k_pack, ew((size_t)HD * HD), dim3(256), 0, stream, fw, wpf, HD, HD);
      gemm(hsb, wpf, fb, nullptr, nullptr, t1b, MROWS, HD, HD, 1, 0);
      gemm(t1b, wpf, fb, nullptr, nullptr, t2b, MROWS, HD, HD, 1, 0);
      gemm(t2b, wpf, fb, nullptr, acc, nullptr, MROWS, HD, HD, 1, (i > 0) ? 1 : 0);
    }
    hipLaunchKernelGGL(k_fracadd, ew((size_t)MROWS * HD), dim3(256), 0, stream, hs, acc);
  }

  /* final LayerNorm -> f32 output */
  hipLaunchKernelGGL(k_layernorm, dim3(MROWS), dim3(256), 0, stream,
                     hs, fln_w, fln_b, (u16*)nullptr, outp);
}
